// MultiHeadSelfAttention_37237366456637
// MI455X (gfx1250) — matrix operand order repaired and checked by reading, not compiled, not run on hardware
//
#include <hip/hip_runtime.h>
#include <stdint.h>

// ---------------------------------------------------------------------------
// MI455X (gfx1250) causal multi-head self-attention, bf16 WMMA pipeline.
// DIM=1024, H=16, Dh=64, B=2, T=2048.
// ---------------------------------------------------------------------------

#define DIMC 1024
#define NHEAD 16
#define DH 64
#define BATCH 2
#define SEQ 2048

typedef __attribute__((ext_vector_type(16))) __bf16 v16bf;
typedef __attribute__((ext_vector_type(2)))  __bf16 v2bf;
typedef __attribute__((ext_vector_type(8)))  float  v8f;

__device__ __forceinline__ v8f wmma_bf16(v16bf a, v16bf b, v8f c) {
  // D = A(16x32 bf16) * B(32x16 bf16) + C(16x16 f32)
  return __builtin_amdgcn_wmma_f32_16x16x32_bf16(
      /*neg_a=*/false, a, /*neg_b=*/false, b,
      /*c_mod=*/(short)0, c, /*reuse_a=*/false, /*reuse_b=*/false);
}

// Single f32 -> bf16 (RNE) via native conversion.
__device__ __forceinline__ uint16_t bf16u(float x) {
  return __builtin_bit_cast(uint16_t, (__bf16)x);
}
// Packed pair -> one 32-bit word (low = a, high = b).
__device__ __forceinline__ uint32_t pkbf16(float a, float b) {
#if __has_builtin(__builtin_amdgcn_cvt_pk_bf16_f32)
  v2bf p = __builtin_amdgcn_cvt_pk_bf16_f32(a, b);
#else
  v2bf p = {(__bf16)a, (__bf16)b};
#endif
  return __builtin_bit_cast(uint32_t, p);
}

union FragU { v16bf v; float4 f[2]; uint4 u[2]; };

// A-matrix fragment (16x32 bf16): lane holds row m=lane&15.
// half=lane>>4: VGPR0-3 = K in [half*8, half*8+8), VGPR4-7 = K in [16+half*8,..).
__device__ __forceinline__ v16bf ld_fragA(const uint16_t* p) {
  FragU u;
  u.f[0] = *(const float4*)(p);
  u.f[1] = *(const float4*)(p + 16);
  return u.v;
}

// B-matrix fragment (32x16 bf16): lane holds col n=lane&15.
// half=lane>>4: VGPR0-3 = K in [half*8, half*8+8), VGPR4-7 = K in [16+half*8,..).
__device__ __forceinline__ v16bf ld_fragB(const uint16_t* p) {
  FragU u;
  u.f[0] = *(const float4*)(p);
  u.f[1] = *(const float4*)(p + 16);
  return u.v;
}

__device__ __forceinline__ float rmax16(float v) {
  v = fmaxf(v, __shfl_xor(v, 1, 32));
  v = fmaxf(v, __shfl_xor(v, 2, 32));
  v = fmaxf(v, __shfl_xor(v, 4, 32));
  v = fmaxf(v, __shfl_xor(v, 8, 32));
  return v;
}
__device__ __forceinline__ float rsum16(float v) {
  v += __shfl_xor(v, 1, 32);
  v += __shfl_xor(v, 2, 32);
  v += __shfl_xor(v, 4, 32);
  v += __shfl_xor(v, 8, 32);
  return v;
}

// gfx1250 async global->LDS copy, 16B per lane. LDS dest address comes from a
// VGPR (low 32 bits of the flat LDS address); tracked by ASYNCcnt.
__device__ __forceinline__ void async_ld_b128(const uint16_t* lds_dst,
                                              const uint16_t* gsrc) {
  uint32_t l = (uint32_t)(uintptr_t)lds_dst;
  uint64_t g = (uint64_t)(uintptr_t)gsrc;
  asm volatile("global_load_async_to_lds_b128 %0, %1, off"
               :: "v"(l), "v"(g) : "memory");
}
__device__ __forceinline__ void wait_async0() {
  asm volatile("s_wait_asynccnt 0" ::: "memory");
}

// ---------------------------------------------------------------------------
// Kernel 1: qkv = x @ w_qkv + b_qkv  ->  scatter to q/k/v bf16 [B,H,T,Dh]
// M=4096 K=1024 N=3072. Block tile 128x128, 8 waves (2x4), wave tile 64x32.
// ---------------------------------------------------------------------------
__global__ __launch_bounds__(256) void qkv_gemm_kernel(
    const float* __restrict__ x, const float* __restrict__ w,
    const float* __restrict__ bias,
    uint16_t* __restrict__ qw, uint16_t* __restrict__ kw,
    uint16_t* __restrict__ vw) {
  __shared__ uint16_t As[128 * 40];   // A tile, row-major along K, stride 40
  __shared__ uint16_t Bs[128 * 40];   // B tile transposed: [n][k], stride 40

  const int tid  = threadIdx.x;
  const int lane = tid & 31;
  const int wave = tid >> 5;
  const int ln = lane & 15, hl = lane >> 4;
  const int Mbase = blockIdx.y * 128;
  const int Nbase = blockIdx.x * 128;
  const int wm = wave >> 2, wn = wave & 3;

  v8f acc[4][2];
  v8f zero = {0.f, 0.f, 0.f, 0.f, 0.f, 0.f, 0.f, 0.f};
#pragma unroll
  for (int mt = 0; mt < 4; ++mt)
#pragma unroll
    for (int nt = 0; nt < 2; ++nt) acc[mt][nt] = zero;

  for (int k0 = 0; k0 < DIMC; k0 += 32) {
    // --- A tile: 128x32 f32 -> bf16, packed b32 stores ---
    {
      int c4 = tid & 7;          // float4 within 32-col row
      int r0 = tid >> 3;         // 0..31
#pragma unroll
      for (int rr = 0; rr < 4; ++rr) {
        int r = r0 + rr * 32;
        float4 v = *(const float4*)&x[(size_t)(Mbase + r) * DIMC + k0 + c4 * 4];
        uint32_t* d = (uint32_t*)&As[r * 40 + c4 * 4];
        d[0] = pkbf16(v.x, v.y);
        d[1] = pkbf16(v.z, v.w);
      }
    }
    // --- B tile: 32(K) x 128(N) f32 -> bf16, transposed to [n][k] ---
    {
      int c4 = tid & 31;         // float4 across 128 cols
      int kr = tid >> 5;         // 0..7
#pragma unroll
      for (int kk = 0; kk < 4; ++kk) {
        int k = kr + kk * 8;
        float4 v = *(const float4*)&w[(size_t)(k0 + k) * 3072 + Nbase + c4 * 4];
        Bs[(c4 * 4 + 0) * 40 + k] = bf16u(v.x);
        Bs[(c4 * 4 + 1) * 40 + k] = bf16u(v.y);
        Bs[(c4 * 4 + 2) * 40 + k] = bf16u(v.z);
        Bs[(c4 * 4 + 3) * 40 + k] = bf16u(v.w);
      }
    }
    __syncthreads();

    v16bf afr[4], bfr[2];
#pragma unroll
    for (int mt = 0; mt < 4; ++mt)
      afr[mt] = ld_fragA(&As[(wm * 64 + mt * 16 + ln) * 40 + hl * 8]);
#pragma unroll
    for (int nt = 0; nt < 2; ++nt)
      bfr[nt] = ld_fragB(&Bs[(wn * 32 + nt * 16 + ln) * 40 + hl * 8]);
#pragma unroll
    for (int mt = 0; mt < 4; ++mt)
#pragma unroll
      for (int nt = 0; nt < 2; ++nt)
        acc[mt][nt] = wmma_bf16(afr[mt], bfr[nt], acc[mt][nt]);
    __syncthreads();
  }

  // Epilogue: +bias, scale Q by 1/sqrt(Dh), scatter to [B,H,T,Dh] bf16.
#pragma unroll
  for (int mt = 0; mt < 4; ++mt) {
#pragma unroll
    for (int nt = 0; nt < 2; ++nt) {
      int n_g = Nbase + wn * 32 + nt * 16 + ln;
      float bv = bias[n_g];
      int which = n_g >> 10;           // 0=q 1=k 2=v
      int cc = n_g & 1023;
      int h = cc >> 6, d = cc & 63;
      uint16_t* dst = (which == 0) ? qw : ((which == 1) ? kw : vw);
      float scl = (which == 0) ? 0.125f : 1.0f;   // fold 1/sqrt(64) into Q
#pragma unroll
      for (int r = 0; r < 8; ++r) {
        int m = Mbase + wm * 64 + mt * 16 + r + hl * 8;
        int bI = m >> 11, t = m & 2047;
        float val = (acc[mt][nt][r] + bv) * scl;
        dst[((size_t)(bI * NHEAD + h) * SEQ + t) * DH + d] = bf16u(val);
      }
    }
  }
}

// ---------------------------------------------------------------------------
// Kernel 2: causal flash attention. One block per (b, h, 256-q-row chunk).
// 8 waves x 32 q-rows each; key tiles of 64; online softmax in f32.
// K tile loaded via gfx1250 async global->LDS DMA.
// ---------------------------------------------------------------------------
__global__ __launch_bounds__(256) void attn_kernel(
    const uint16_t* __restrict__ qw, const uint16_t* __restrict__ kw,
    const uint16_t* __restrict__ vw, uint16_t* __restrict__ attn) {
  __shared__ uint16_t Ks[64 * 80];        // [key][d]     (B-frag source for S)
  __shared__ uint16_t Vs[64 * 80];        // [d][key]     (B-frag source for PV)
  __shared__ uint16_t Ps[8 * 32 * 80];    // per-wave P   (A-frag source for PV)

  const int tid  = threadIdx.x;
  const int lane = tid & 31;
  const int wave = tid >> 5;
  const int ln = lane & 15, hl = lane >> 4;

  const int bid = blockIdx.x;
  const int qc = bid & 7;
  const int h  = (bid >> 3) & 15;
  const int b  = bid >> 7;
  const size_t headoff = (size_t)(b * NHEAD + h) * SEQ * DH;
  const uint16_t* Q = qw + headoff;
  const uint16_t* K = kw + headoff;
  const uint16_t* V = vw + headoff;
  const int qbase = qc * 256 + wave * 32;

  // Q fragments resident in VGPRs: 2 m-tiles x 2 K-steps (Dh=64).
  v16bf qf[2][2];
#pragma unroll
  for (int mt = 0; mt < 2; ++mt)
#pragma unroll
    for (int ks = 0; ks < 2; ++ks) {
      const uint16_t* p = &Q[(size_t)(qbase + mt * 16 + ln) * DH + ks * 32 + hl * 8];
      FragU u;
      u.u[0] = *(const uint4*)(p);
      u.u[1] = *(const uint4*)(p + 16);
      qf[mt][ks] = u.v;
    }

  v8f o[2][4];
  v8f zero = {0.f, 0.f, 0.f, 0.f, 0.f, 0.f, 0.f, 0.f};
#pragma unroll
  for (int mt = 0; mt < 2; ++mt)
#pragma unroll
    for (int dt = 0; dt < 4; ++dt) o[mt][dt] = zero;
  float rowM[2][8], rowL[2][8];
#pragma unroll
  for (int mt = 0; mt < 2; ++mt)
#pragma unroll
    for (int r = 0; r < 8; ++r) { rowM[mt][r] = -1e30f; rowL[mt][r] = 0.f; }

  const int ktmax = qc * 4 + 3;   // causal: keys up to block's last q row
  const int wb = wave * 32 * 80;  // this wave's P region

  for (int kt = 0; kt <= ktmax; ++kt) {
    const int kb = kt * 64;
    // K tile: straight strided copy -> async DMA, 16B per lane.
    // V tile: element transpose into [d][key] (manual).
    for (int c = tid; c < 512; c += 256) {
      int row = c >> 3, c8 = (c & 7) * 8;
      async_ld_b128(&Ks[row * 80 + c8], &K[(size_t)(kb + row) * DH + c8]);
      union { uint4 u; uint16_t s[8]; } tv;
      tv.u = *(const uint4*)&V[(size_t)(kb + row) * DH + c8];
#pragma unroll
      for (int j = 0; j < 8; ++j) Vs[(c8 + j) * 80 + row] = tv.s[j];
    }
    wait_async0();
    __syncthreads();

    if (kb <= qbase + 31) {
      // ---- S = Q . K^T  (2x4 tiles of 16x16, Dh in 2 WMMA K-steps) ----
      v8f s[2][4];
#pragma unroll
      for (int mt = 0; mt < 2; ++mt)
#pragma unroll
        for (int nt = 0; nt < 4; ++nt) s[mt][nt] = zero;
#pragma unroll
      for (int ks = 0; ks < 2; ++ks) {
        v16bf kf[4];
#pragma unroll
        for (int nt = 0; nt < 4; ++nt)
          kf[nt] = ld_fragB(&Ks[(nt * 16 + ln) * 80 + ks * 32 + hl * 8]);
#pragma unroll
        for (int mt = 0; mt < 2; ++mt)
#pragma unroll
          for (int nt = 0; nt < 4; ++nt)
            s[mt][nt] = wmma_bf16(qf[mt][ks], kf[nt], s[mt][nt]);
      }

      // ---- causal mask + online softmax (per-row, 16-lane reductions) ----
#pragma unroll
      for (int mt = 0; mt < 2; ++mt) {
#pragma unroll
        for (int r = 0; r < 8; ++r) {
          int qg = qbase + mt * 16 + r + hl * 8;
          float mloc = -1e30f;
#pragma unroll
          for (int nt = 0; nt < 4; ++nt) {
            int kg = kb + nt * 16 + ln;
            float sv = s[mt][nt][r];
            if (kg > qg) sv = -1e30f;
            s[mt][nt][r] = sv;
            mloc = fmaxf(mloc, sv);
          }
          mloc = rmax16(mloc);
          float mnew = fmaxf(rowM[mt][r], mloc);
          float scl = __expf(rowM[mt][r] - mnew);
          rowM[mt][r] = mnew;
          float psum = 0.f;
#pragma unroll
          for (int nt = 0; nt < 4; ++nt) {
            float p = __expf(s[mt][nt][r] - mnew);
            s[mt][nt][r] = p;
            psum += p;
          }
          psum = rsum16(psum);
          rowL[mt][r] = rowL[mt][r] * scl + psum;
#pragma unroll
          for (int dt = 0; dt < 4; ++dt) o[mt][dt][r] *= scl;
        }
      }

      // ---- P (C-layout) -> LDS -> A-layout fragments ----
#pragma unroll
      for (int mt = 0; mt < 2; ++mt)
#pragma unroll
        for (int nt = 0; nt < 4; ++nt)
#pragma unroll
          for (int r = 0; r < 8; ++r)
            Ps[wb + (mt * 16 + r + hl * 8) * 80 + nt * 16 + ln] =
                bf16u(s[mt][nt][r]);

      // ---- O += P . V  (2x4 tiles, 64 keys in 2 WMMA K-steps) ----
#pragma unroll
      for (int ks = 0; ks < 2; ++ks) {
        v16bf pf[2], vf[4];
#pragma unroll
        for (int mt = 0; mt < 2; ++mt)
          pf[mt] = ld_fragA(&Ps[wb + (mt * 16 + ln) * 80 + ks * 32 + hl * 8]);
#pragma unroll
        for (int dt = 0; dt < 4; ++dt)
          vf[dt] = ld_fragB(&Vs[(dt * 16 + ln) * 80 + ks * 32 + hl * 8]);
#pragma unroll
        for (int mt = 0; mt < 2; ++mt)
#pragma unroll
          for (int dt = 0; dt < 4; ++dt)
            o[mt][dt] = wmma_bf16(pf[mt], vf[dt], o[mt][dt]);
      }
    }
    __syncthreads();
  }

  // Normalize and store attention output as bf16 [B, T, DIM] (head-interleaved)
#pragma unroll
  for (int mt = 0; mt < 2; ++mt)
#pragma unroll
    for (int r = 0; r < 8; ++r) {
      float inv = 1.0f / rowL[mt][r];
      int t = qbase + mt * 16 + r + hl * 8;
#pragma unroll
      for (int dt = 0; dt < 4; ++dt) {
        int col = h * DH + dt * 16 + ln;
        attn[(size_t)(b * SEQ + t) * DIMC + col] = bf16u(o[mt][dt][r] * inv);
      }
    }
}

// ---------------------------------------------------------------------------
// Kernel 3: out = attn_bf16 @ w_proj + b_proj  (f32 out). M=4096 K=1024 N=1024.
// A tile loaded via gfx1250 async global->LDS DMA (pure bf16 copy).
// ---------------------------------------------------------------------------
__global__ __launch_bounds__(256) void proj_gemm_kernel(
    const uint16_t* __restrict__ a, const float* __restrict__ w,
    const float* __restrict__ bias, float* __restrict__ out) {
  __shared__ uint16_t As[128 * 40];
  __shared__ uint16_t Bs[128 * 40];

  const int tid  = threadIdx.x;
  const int lane = tid & 31;
  const int wave = tid >> 5;
  const int ln = lane & 15, hl = lane >> 4;
  const int Mbase = blockIdx.y * 128;
  const int Nbase = blockIdx.x * 128;
  const int wm = wave >> 2, wn = wave & 3;

  v8f acc[4][2];
  v8f zero = {0.f, 0.f, 0.f, 0.f, 0.f, 0.f, 0.f, 0.f};
#pragma unroll
  for (int mt = 0; mt < 4; ++mt)
#pragma unroll
    for (int nt = 0; nt < 2; ++nt) acc[mt][nt] = zero;

  for (int k0 = 0; k0 < DIMC; k0 += 32) {
    // A tile already bf16: async 16B copies into LDS.
    for (int c = tid; c < 512; c += 256) {
      int row = c >> 2, c8 = (c & 3) * 8;
      async_ld_b128(&As[row * 40 + c8],
                    &a[(size_t)(Mbase + row) * DIMC + k0 + c8]);
    }
    // B tile: f32 -> bf16, transposed into [n][k].
    {
      int c4 = tid & 31;
      int kr = tid >> 5;
#pragma unroll
      for (int kk = 0; kk < 4; ++kk) {
        int k = kr + kk * 8;
        float4 v = *(const float4*)&w[(size_t)(k0 + k) * DIMC + Nbase + c4 * 4];
        Bs[(c4 * 4 + 0) * 40 + k] = bf16u(v.x);
        Bs[(c4 * 4 + 1) * 40 + k] = bf16u(v.y);
        Bs[(c4 * 4 + 2) * 40 + k] = bf16u(v.z);
        Bs[(c4 * 4 + 3) * 40 + k] = bf16u(v.w);
      }
    }
    wait_async0();
    __syncthreads();

    v16bf afr[4], bfr[2];
#pragma unroll
    for (int mt = 0; mt < 4; ++mt)
      afr[mt] = ld_fragA(&As[(wm * 64 + mt * 16 + ln) * 40 + hl * 8]);
#pragma unroll
    for (int nt = 0; nt < 2; ++nt)
      bfr[nt] = ld_fragB(&Bs[(wn * 32 + nt * 16 + ln) * 40 + hl * 8]);
#pragma unroll
    for (int mt = 0; mt < 4; ++mt)
#pragma unroll
      for (int nt = 0; nt < 2; ++nt)
        acc[mt][nt] = wmma_bf16(afr[mt], bfr[nt], acc[mt][nt]);
    __syncthreads();
  }

#pragma unroll
  for (int mt = 0; mt < 4; ++mt)
#pragma unroll
    for (int nt = 0; nt < 2; ++nt) {
      int n = Nbase + wn * 32 + nt * 16 + ln;
      float bv = bias[n];
#pragma unroll
      for (int r = 0; r < 8; ++r) {
        int m = Mbase + wm * 64 + mt * 16 + r + hl * 8;
        out[(size_t)m * DIMC + n] = acc[mt][nt][r] + bv;
      }
    }
}

// ---------------------------------------------------------------------------
extern "C" void kernel_launch(void* const* d_in, const int* in_sizes, int n_in,
                              void* d_out, int out_size, void* d_ws,
                              size_t ws_size, hipStream_t stream) {
  const float* x      = (const float*)d_in[0];
  const float* w_qkv  = (const float*)d_in[1];
  const float* b_qkv  = (const float*)d_in[2];
  const float* w_proj = (const float*)d_in[3];
  const float* b_proj = (const float*)d_in[4];
  float* out = (float*)d_out;

  const size_t per = (size_t)BATCH * NHEAD * SEQ * DH;   // 4,194,304 elems
  uint16_t* qw   = (uint16_t*)d_ws;
  uint16_t* kw   = qw + per;
  uint16_t* vw   = kw + per;
  uint16_t* attn = vw + per;                             // [B,T,DIM] bf16

  // 1) QKV projection: grid (3072/128, 4096/128)
  qkv_gemm_kernel<<<dim3(24, 32), 256, 0, stream>>>(x, w_qkv, b_qkv, qw, kw, vw);
  // 2) Flash attention: B*H*(T/256) = 256 blocks
  attn_kernel<<<dim3(256), 256, 0, stream>>>(qw, kw, vw, attn);
  // 3) Output projection: grid (1024/128, 4096/128)
  proj_gemm_kernel<<<dim3(8, 32), 256, 0, stream>>>(attn, w_proj, b_proj, out);
}
